// MixtralModel_33535104647334
// MI455X (gfx1250) — matrix operand order repaired and checked by reading, not compiled, not run on hardware
//
#include <hip/hip_runtime.h>
#include <hip/hip_bf16.h>
#include <math.h>

// ---------------- model dims ----------------
constexpr int SEQ  = 512;
constexpr int HDIM = 1024;
constexpr int NH   = 16;
constexpr int NKV  = 8;
constexpr int HD   = 64;
constexpr int NE   = 8;     // experts
constexpr int FF   = 2048;
constexpr int NL   = 2;
constexpr int QKVD = (NH + 2 * NKV) * HD;  // 2048
constexpr float EPSV  = 1e-5f;
constexpr float SCALE = 0.125f;            // 64^-0.5

typedef __attribute__((ext_vector_type(16))) __bf16    v16bf;
typedef __attribute__((ext_vector_type(8)))  float     v8f;
typedef __attribute__((ext_vector_type(8)))  unsigned  v8u;

// Pack two fp32 into one dword of two bf16 (round-to-nearest, ties-up):
// one v_add per element + one v_perm_b32 per pair -> 1.5 VALU/elem.
__device__ __forceinline__ unsigned pack_bf16(float x, float y) {
  unsigned ux = __builtin_bit_cast(unsigned, x) + 0x8000u;
  unsigned uy = __builtin_bit_cast(unsigned, y) + 0x8000u;
  // D bytes = { uy[3], uy[2], ux[3], ux[2] }  (perm: {src0=hi, src1=lo})
  return __builtin_amdgcn_perm(uy, ux, 0x07060302u);
}

// A fragment: 16x32 bf16. lanes 0-15 -> K k0+0..15 ; lanes 16-31 -> K k0+16..31
__device__ __forceinline__ v16bf load_frag_a(const float* __restrict__ p) {
  const float4* p0 = (const float4*)p;
  const float4* p1 = (const float4*)(p + 8);
  float4 x0 = p0[0], x1 = p0[1], x2 = p1[0], x3 = p1[1];
  v8u r;
  r[0] = pack_bf16(x0.x, x0.y); r[1] = pack_bf16(x0.z, x0.w);
  r[2] = pack_bf16(x1.x, x1.y); r[3] = pack_bf16(x1.z, x1.w);
  r[4] = pack_bf16(x2.x, x2.y); r[5] = pack_bf16(x2.z, x2.w);
  r[6] = pack_bf16(x3.x, x3.y); r[7] = pack_bf16(x3.z, x3.w);
  return __builtin_bit_cast(v16bf, r);
}

// B fragment: 32x16 bf16. lanes 0-15 -> K k0+0..15 ; lanes 16-31 -> K k0+16..31
__device__ __forceinline__ v16bf load_frag_b(const float* __restrict__ p) {
  const float4* q = (const float4*)p;
  float4 x0 = q[0], x1 = q[1], x2 = q[2], x3 = q[3];
  v8u r;
  r[0] = pack_bf16(x0.x, x0.y); r[1] = pack_bf16(x0.z, x0.w);
  r[2] = pack_bf16(x1.x, x1.y); r[3] = pack_bf16(x1.z, x1.w);
  r[4] = pack_bf16(x2.x, x2.y); r[5] = pack_bf16(x2.z, x2.w);
  r[6] = pack_bf16(x3.x, x3.y); r[7] = pack_bf16(x3.z, x3.w);
  return __builtin_bit_cast(v16bf, r);
}

#define WMMA_BF16(a, b, c) \
  __builtin_amdgcn_wmma_f32_16x16x32_bf16(false, (a), false, (b), (short)0, (c), false, false)

// ---------------- embedding gather ----------------
__global__ void embed_kernel(const int* __restrict__ ids,
                             const float* __restrict__ embed,
                             float* __restrict__ h) {
  int t = blockIdx.x;
  size_t row = (size_t)ids[t] * HDIM;
  for (int j = threadIdx.x; j < HDIM; j += blockDim.x)
    h[(size_t)t * HDIM + j] = embed[row + j];
}

// ---------------- RMSNorm (one block per token) ----------------
__global__ void rmsnorm_kernel(const float* __restrict__ x,
                               const float* __restrict__ w,
                               float* __restrict__ y) {
  __shared__ float red[256];
  int t = blockIdx.x, tid = threadIdx.x;
  const float* xr = x + (size_t)t * HDIM;
  float ss = 0.f;
  for (int j = tid; j < HDIM; j += 256) { float v = xr[j]; ss += v * v; }
  red[tid] = ss; __syncthreads();
  for (int s = 128; s > 0; s >>= 1) { if (tid < s) red[tid] += red[tid + s]; __syncthreads(); }
  float inv = rsqrtf(red[0] / (float)HDIM + EPSV);
  __syncthreads();
  for (int j = tid; j < HDIM; j += 256)
    y[(size_t)t * HDIM + j] = xr[j] * inv * w[j];
}

// ---------------- WMMA bf16 GEMM: C[M,N] = A[M,KC] * B[NC,KC]^T ----------------
// block = 256 threads = 8 waves in 4(M) x 2(N); each wave owns a 32(M)x64(N)
// output tile (2x4 accumulators -> 8 WMMAs per K-step from 6 fragment loads).
// Block tile 128(M) x 128(N). Requires M%128==0, NC%128==0, KC%32==0.
// NC/KC are compile-time so all row strides fold into load/store immediates
// (keeps live pointers to 2 base addresses -> no VGPR spill in the hot loop).
template <int NC, int KC>
__global__ void gemm_bf16_wmma(const float* __restrict__ A,
                               const float* __restrict__ B,
                               float* __restrict__ C) {
  unsigned tid  = threadIdx.x;
  unsigned lane = tid & 31u;
  unsigned wv   = tid >> 5;
  unsigned mt   = wv & 3u;
  unsigned nt   = wv >> 2;
  unsigned m0   = blockIdx.y * 128u + mt * 32u;
  unsigned n0   = blockIdx.x * 128u + nt * 64u;
  unsigned grp  = lane >> 4;     // 0 or 1 (lane half)
  unsigned l16  = lane & 15u;

  // One base pointer per matrix; all other rows via constant immediates.
  const float* ap = A + (size_t)(m0 + l16) * KC + grp * 16;
  const float* bp = B + (size_t)(n0 + l16) * KC + grp * 16;

  v8f c00 = {}, c01 = {}, c02 = {}, c03 = {};
  v8f c10 = {}, c11 = {}, c12 = {}, c13 = {};
  for (int k0 = 0; k0 < KC; k0 += 32, ap += 32, bp += 32) {
    v16bf a0 = load_frag_a(ap);
    v16bf a1 = load_frag_a(ap + 16 * KC);
    v16bf b0 = load_frag_b(bp);
    v16bf b1 = load_frag_b(bp + 16 * KC);
    v16bf b2 = load_frag_b(bp + 32 * KC);
    v16bf b3 = load_frag_b(bp + 48 * KC);
    c00 = WMMA_BF16(a0, b0, c00);
    c01 = WMMA_BF16(a0, b1, c01);
    c02 = WMMA_BF16(a0, b2, c02);
    c03 = WMMA_BF16(a0, b3, c03);
    c10 = WMMA_BF16(a1, b0, c10);
    c11 = WMMA_BF16(a1, b1, c11);
    c12 = WMMA_BF16(a1, b2, c12);
    c13 = WMMA_BF16(a1, b3, c13);
  }
  // C/D layout: VGPR r -> M = tileM + r + 8*grp ; N = tileN + (lane&15)
  float* cp = C + (size_t)(m0 + 8 * grp) * NC + n0 + l16;
#pragma unroll
  for (int r = 0; r < 8; ++r) {
    cp[r * NC]                 = c00[r];
    cp[r * NC + 16]            = c01[r];
    cp[r * NC + 32]            = c02[r];
    cp[r * NC + 48]            = c03[r];
    cp[(r + 16) * NC]          = c10[r];
    cp[(r + 16) * NC + 16]     = c11[r];
    cp[(r + 16) * NC + 32]     = c12[r];
    cp[(r + 16) * NC + 48]     = c13[r];
  }
}

// ---------------- RoPE (neox rotate-half), in-place on qkv buffer ----------------
// heads 0..15 = q heads ; heads 16..23 = kv k heads
__global__ void rope_kernel(float* __restrict__ qkv, const int* __restrict__ pos) {
  int idx = blockIdx.x * blockDim.x + threadIdx.x;
  if (idx >= SEQ * 24 * 32) return;
  int i  = idx & 31;
  int hh = (idx >> 5) % 24;
  int s  = idx / (24 * 32);
  float p    = (float)pos[s];
  float freq = p * expf(-logf(10000.0f) * (2.0f * (float)i / (float)HD));
  float c = cosf(freq), sn = sinf(freq);
  float* base = qkv + (size_t)s * QKVD + (hh < NH ? hh * HD : NH * HD + (hh - NH) * HD);
  float x1 = base[i], x2 = base[i + 32];
  base[i]      = x1 * c - x2 * sn;
  base[i + 32] = x2 * c + x1 * sn;
}

// ---------------- fused causal attention: one block per (query, head) ----------------
__global__ void attn_kernel(const float* __restrict__ qkv, float* __restrict__ out) {
  __shared__ float qs[HD];
  __shared__ float sc[SEQ];
  __shared__ float red[256];
  __shared__ float part[256];
  int q = blockIdx.x, hh = blockIdx.y, tid = threadIdx.x;
  int kvh = hh >> 1;                       // GQA rep = NH/NKV = 2
  if (tid < HD) qs[tid] = qkv[(size_t)q * QKVD + hh * HD + tid];
  __syncthreads();
  int nk = q + 1;
  for (int k = tid; k < nk; k += 256) {
    const float* kv = qkv + (size_t)k * QKVD + NH * HD + kvh * HD;
    float d = 0.f;
#pragma unroll
    for (int j = 0; j < HD; ++j) d += qs[j] * kv[j];
    sc[k] = d * SCALE;
  }
  __syncthreads();
  float lm = -3.4e38f;
  for (int k = tid; k < nk; k += 256) lm = fmaxf(lm, sc[k]);
  red[tid] = lm; __syncthreads();
  for (int s = 128; s > 0; s >>= 1) { if (tid < s) red[tid] = fmaxf(red[tid], red[tid + s]); __syncthreads(); }
  float mx = red[0]; __syncthreads();
  float ls = 0.f;
  for (int k = tid; k < nk; k += 256) { float e = __expf(sc[k] - mx); sc[k] = e; ls += e; }
  red[tid] = ls; __syncthreads();
  for (int s = 128; s > 0; s >>= 1) { if (tid < s) red[tid] += red[tid + s]; __syncthreads(); }
  float inv = 1.0f / red[0];
  int d = tid & 63, c = tid >> 6;          // 4 k-chunks x 64 dims
  float acc = 0.f;
  for (int k = c; k < nk; k += 4)
    acc += sc[k] * qkv[(size_t)k * QKVD + (NH + NKV) * HD + kvh * HD + d];
  part[tid] = acc; __syncthreads();
  if (tid < 64)
    out[(size_t)q * (NH * HD) + hh * HD + d] =
        (part[tid] + part[tid + 64] + part[tid + 128] + part[tid + 192]) * inv;
}

// ---------------- elementwise helpers ----------------
__global__ void add_kernel(float* __restrict__ dst, const float* __restrict__ src, int n) {
  int i = blockIdx.x * blockDim.x + threadIdx.x;
  if (i < n) dst[i] += src[i];
}

__global__ void silu_mul_kernel(float* __restrict__ g, const float* __restrict__ u, int n) {
  int i = blockIdx.x * blockDim.x + threadIdx.x;
  if (i < n) { float x = g[i]; g[i] = (x / (1.0f + __expf(-x))) * u[i]; }
}

__global__ void scaled_accum_kernel(float* __restrict__ h, const float* __restrict__ y,
                                    const float* __restrict__ comb, int e) {
  int idx = blockIdx.x * blockDim.x + threadIdx.x;   // SEQ*HDIM
  int t = idx >> 10, j = idx & 1023;
  float w = comb[t * NE + e];
  h[(size_t)t * HDIM + j] += w * y[(size_t)t * HDIM + j];
}

// ---------------- MoE gating: one block per token, wave per expert ----------------
__global__ void gate_kernel(const float* __restrict__ x, const float* __restrict__ gw,
                            float* __restrict__ comb) {
  __shared__ float lg[NE];
  int t = blockIdx.x, tid = threadIdx.x;
  int lane = tid & 31, e = tid >> 5;
  const float* xr = x + (size_t)t * HDIM;
  float acc = 0.f;
  for (int j = lane; j < HDIM; j += 32) acc += xr[j] * gw[(size_t)e * HDIM + j];
  for (int m = 16; m > 0; m >>= 1) acc += __shfl_xor(acc, m, 32);
  if (lane == 0) lg[e] = acc;
  __syncthreads();
  if (tid == 0) {
    float mx = lg[0];
    for (int i = 1; i < NE; ++i) mx = fmaxf(mx, lg[i]);
    float p[NE]; float ssum = 0.f;
    for (int i = 0; i < NE; ++i) { p[i] = __expf(lg[i] - mx); ssum += p[i]; }
    for (int i = 0; i < NE; ++i) p[i] /= ssum;
    int i0 = 0;
    for (int i = 1; i < NE; ++i) if (p[i] > p[i0]) i0 = i;     // lowest index on tie
    int i1 = (i0 == 0) ? 1 : 0;
    for (int i = 0; i < NE; ++i) if (i != i0 && p[i] > p[i1]) i1 = i;
    float sw = p[i0] + p[i1];
    for (int i = 0; i < NE; ++i) comb[t * NE + i] = 0.f;
    comb[t * NE + i0] = p[i0] / sw;
    comb[t * NE + i1] = p[i1] / sw;
  }
}

// ---------------- host-side orchestration ----------------
extern "C" void kernel_launch(void* const* d_in, const int* in_sizes, int n_in,
                              void* d_out, int out_size, void* d_ws, size_t ws_size,
                              hipStream_t stream) {
  (void)in_sizes; (void)n_in; (void)out_size; (void)ws_size;
  const int*   ids   = (const int*)d_in[0];
  const int*   pos   = (const int*)d_in[1];
  const float* emb   = (const float*)d_in[2];
  const float* ln1   = (const float*)d_in[3];
  const float* qkvw  = (const float*)d_in[4];
  const float* ow    = (const float*)d_in[5];
  const float* ln2   = (const float*)d_in[6];
  const float* gatew = (const float*)d_in[7];
  const float* wg    = (const float*)d_in[8];
  const float* wu    = (const float*)d_in[9];
  const float* wd    = (const float*)d_in[10];
  const float* normw = (const float*)d_in[11];

  float* ws   = (float*)d_ws;                 // ~21 MB used
  float* h    = ws;                           // [SEQ,HDIM]
  float* hn   = h    + (size_t)SEQ * HDIM;    // [SEQ,HDIM]
  float* qkv  = hn   + (size_t)SEQ * HDIM;    // [SEQ,QKVD]
  float* att  = qkv  + (size_t)SEQ * QKVD;    // [SEQ,HDIM]
  float* tmp  = att  + (size_t)SEQ * HDIM;    // [SEQ,HDIM]  (o-proj out / expert y)
  float* g    = tmp  + (size_t)SEQ * HDIM;    // [SEQ,FF]
  float* u    = g    + (size_t)SEQ * FF;      // [SEQ,FF]
  float* comb = u    + (size_t)SEQ * FF;      // [SEQ,NE]

  dim3 blk(256);
  dim3 grid_n2048(QKVD / 128, SEQ / 128);   // N=2048 tiles
  dim3 grid_n1024(HDIM / 128, SEQ / 128);   // N=1024 tiles

  embed_kernel<<<SEQ, blk, 0, stream>>>(ids, emb, h);

  for (int l = 0; l < NL; ++l) {
    // ---- attention block ----
    rmsnorm_kernel<<<SEQ, blk, 0, stream>>>(h, ln1 + (size_t)l * HDIM, hn);
    gemm_bf16_wmma<QKVD, HDIM><<<grid_n2048, blk, 0, stream>>>(
        hn, qkvw + (size_t)l * QKVD * HDIM, qkv);
    rope_kernel<<<(SEQ * 24 * 32 + 255) / 256, blk, 0, stream>>>(qkv, pos);
    attn_kernel<<<dim3(SEQ, NH), blk, 0, stream>>>(qkv, att);
    gemm_bf16_wmma<HDIM, NH * HD><<<grid_n1024, blk, 0, stream>>>(
        att, ow + (size_t)l * HDIM * (NH * HD), tmp);
    add_kernel<<<(SEQ * HDIM + 255) / 256, blk, 0, stream>>>(h, tmp, SEQ * HDIM);

    // ---- MoE block ----
    rmsnorm_kernel<<<SEQ, blk, 0, stream>>>(h, ln2 + (size_t)l * HDIM, hn);
    gate_kernel<<<SEQ, blk, 0, stream>>>(hn, gatew + (size_t)l * NE * HDIM, comb);
    for (int e = 0; e < NE; ++e) {
      const float* wge = wg + ((size_t)l * NE + e) * (size_t)FF * HDIM;
      const float* wue = wu + ((size_t)l * NE + e) * (size_t)FF * HDIM;
      const float* wde = wd + ((size_t)l * NE + e) * (size_t)HDIM * FF;
      gemm_bf16_wmma<FF, HDIM><<<grid_n2048, blk, 0, stream>>>(hn, wge, g);
      gemm_bf16_wmma<FF, HDIM><<<grid_n2048, blk, 0, stream>>>(hn, wue, u);
      silu_mul_kernel<<<(SEQ * FF + 255) / 256, blk, 0, stream>>>(g, u, SEQ * FF);
      gemm_bf16_wmma<HDIM, FF><<<grid_n1024, blk, 0, stream>>>(g, wde, tmp);
      scaled_accum_kernel<<<(SEQ * HDIM + 255) / 256, blk, 0, stream>>>(h, tmp, comb, e);
    }
  }
  rmsnorm_kernel<<<SEQ, blk, 0, stream>>>(h, normw, (float*)d_out);
}
